// FigLinear_35407710388649
// MI455X (gfx1250) — matrix operand order repaired and checked by reading, not compiled, not run on hardware
//
#include <hip/hip_runtime.h>

#define IN_F   4096
#define OUT_F  4096
#define M_TOT  8192        // 4 * 2048 rows
#define BM     128
#define BN     128
#define BK     32
#define LDS_K  40          // BK + 8 pad: keeps 16B alignment, staggers banks
#define KSTEPS (IN_F / BK) // 128 main K-steps (+1 fused LoRA step)

typedef __attribute__((ext_vector_type(16))) __bf16 v16bf;
typedef __attribute__((ext_vector_type(8)))  float  v8f;

__device__ __forceinline__ unsigned short f2bf(float f) {
    unsigned int u = __builtin_bit_cast(unsigned int, f);
    u += 0x7FFFu + ((u >> 16) & 1u);          // round-to-nearest-even
    return (unsigned short)(u >> 16);
}

// CDNA5 async global->LDS copy (16B per lane), tracked by ASYNCcnt.
__device__ __forceinline__ void async_ld16(unsigned ldsOff, const void* g) {
    asm volatile("global_load_async_to_lds_b128 %0, %1, off"
                 :: "v"(ldsOff), "v"((unsigned long long)(size_t)g)
                 : "memory");
}
__device__ __forceinline__ void wait_async0() {
    asm volatile("s_wait_asynccnt 0x0" ::: "memory");
}

// ---------------- Phase 1: staging kernels ----------------

// Dequant INT4 codebook weights -> bf16 W[o][i] (row-major, = B-matrix [n][k]).
// Each int32 element holds one byte: low nibble -> flat 2t, high nibble -> 2t+1.
__global__ void dequant_kernel(const int* __restrict__ indices,
                               const float* __restrict__ codebook,
                               const float* __restrict__ scales,
                               unsigned short* __restrict__ W) {
    int t = blockIdx.x * blockDim.x + threadIdx.x;        // 8,388,608 threads
    unsigned int p = (unsigned int)indices[t];
    float s = scales[t >> 6];                             // group = (2t)/128
    unsigned int lo = p & 15u, hi = (p >> 4) & 15u;
    unsigned int w = (unsigned int)f2bf(codebook[lo] * s)
                   | ((unsigned int)f2bf(codebook[hi] * s) << 16);
    ((unsigned int*)W)[t] = w;
}

// Cast x f32 -> bf16, 8 elems/thread.
__global__ void cast_x_kernel(const float* __restrict__ x,
                              unsigned short* __restrict__ xb) {
    int t = blockIdx.x * blockDim.x + threadIdx.x;        // 4,194,304 threads
    float4 a = ((const float4*)x)[2 * t];
    float4 b = ((const float4*)x)[2 * t + 1];
    uint4 o;
    o.x = (unsigned int)f2bf(a.x) | ((unsigned int)f2bf(a.y) << 16);
    o.y = (unsigned int)f2bf(a.z) | ((unsigned int)f2bf(a.w) << 16);
    o.z = (unsigned int)f2bf(b.x) | ((unsigned int)f2bf(b.y) << 16);
    o.w = (unsigned int)f2bf(b.z) | ((unsigned int)f2bf(b.w) << 16);
    ((uint4*)xb)[t] = o;
}

// t[row][0..15] = x[row] @ lora_A, t[row][16..31] = 0 (K padded to 32 for WMMA).
__global__ void lora_xA_kernel(const float* __restrict__ x,
                               const float* __restrict__ loraA,
                               unsigned short* __restrict__ tL) {
    int g = blockIdx.x * blockDim.x + threadIdx.x;        // 8192*16 threads
    int r = g & 15, row = g >> 4;
    const float* xr = x + (size_t)row * IN_F;
    float acc = 0.f;
#pragma unroll 8
    for (int i = 0; i < IN_F; ++i) acc += xr[i] * loraA[i * 16 + r];
    tL[row * 32 + r]      = f2bf(acc);
    tL[row * 32 + 16 + r] = 0;
}

// WL[o][r] = 2.0 * lora_B[r][o] for r<16, else 0.  (LORA_SCALE = 32/16 = 2)
__global__ void build_WL_kernel(const float* __restrict__ loraB,
                                unsigned short* __restrict__ WL) {
    int g = blockIdx.x * blockDim.x + threadIdx.x;        // 4096*32 threads
    int r = g & 31, o = g >> 5;
    WL[o * 32 + r] = (r < 16) ? f2bf(2.0f * loraB[r * OUT_F + o]) : (unsigned short)0;
}

// ---------------- Phase 2: WMMA GEMM (double-buffered, async copies) ----------------

struct Frag { union { v16bf v; uint4 q[2]; }; };

__global__ __launch_bounds__(256)
void gemm_kernel(const unsigned short* __restrict__ xb,   // [M_TOT][IN_F] bf16 (A, [m][k])
                 const unsigned short* __restrict__ W,    // [OUT_F][IN_F] bf16 (B, [n][k])
                 const unsigned short* __restrict__ tL,   // [M_TOT][32]   bf16 lora A-step
                 const unsigned short* __restrict__ WL,   // [OUT_F][32]   bf16 lora B-step
                 const float* __restrict__ bias,
                 float* __restrict__ out) {
    __shared__ unsigned short As[2][BM][LDS_K];
    __shared__ unsigned short Bs[2][BN][LDS_K];
    const unsigned BUF_BYTES = BM * LDS_K * 2;            // 10240 B per buffer

    const int tid  = threadIdx.x;
    const int lane = tid & 31;
    const int half = lane >> 4;          // 0: lanes 0-15, 1: lanes 16-31
    const int l16  = lane & 15;
    const int wave = tid >> 5;           // 8 waves
    const int wm   = wave & 1;           // 2 along M (64 rows each)
    const int wn   = wave >> 1;          // 4 along N (32 cols each)

    const int mBlock = blockIdx.y * BM;
    const int nBlock = blockIdx.x * BN;

    v8f acc[4][2];
#pragma unroll
    for (int mi = 0; mi < 4; ++mi)
#pragma unroll
        for (int ni = 0; ni < 2; ++ni)
#pragma unroll
            for (int e = 0; e < 8; ++e) acc[mi][ni][e] = 0.f;

    // global->LDS mapping: 256 threads, each moves 16 bf16 (32B) per matrix per tile
    const int ldRow = tid >> 1;          // 0..127
    const int ldCol = (tid & 1) * 16;    // 0 or 16

    // wave-relative LDS byte offsets for this thread's async destinations
    const unsigned aLds = (unsigned)(size_t)&As[0][ldRow][ldCol];
    const unsigned bLds = (unsigned)(size_t)&Bs[0][ldRow][ldCol];

    // per-thread global source pointers (induction: += BK per issued tile)
    const unsigned short* gaT = xb + (size_t)(mBlock + ldRow) * IN_F + ldCol;
    const unsigned short* gbT = W  + (size_t)(nBlock + ldRow) * IN_F + ldCol;

    auto issue_main = [&](int buf) {     // issue current (gaT, gbT) tile, then advance
        unsigned al = aLds + (unsigned)buf * BUF_BYTES;
        unsigned bl = bLds + (unsigned)buf * BUF_BYTES;
        async_ld16(al,      gaT);
        async_ld16(al + 16, gaT + 8);
        async_ld16(bl,      gbT);
        async_ld16(bl + 16, gbT + 8);
        gaT += BK;
        gbT += BK;
    };

    auto compute = [&](int buf) {
        Frag a[4], b[2];
        // A frag, ISA 16-bit A 16x32 layout: q0 = K{0..7 | 8..15}, q1 = K{16..23 | 24..31}
#pragma unroll
        for (int mi = 0; mi < 4; ++mi) {
            const unsigned short* p = &As[buf][wm * 64 + mi * 16 + l16][half * 8];
            a[mi].q[0] = *(const uint4*)p;
            a[mi].q[1] = *(const uint4*)(p + 16);
        }
        // B frag, 32x16 layout: q0 = K{0..7 | 8..15}, q1 = K{16..23 | 24..31} (same as A)
#pragma unroll
        for (int ni = 0; ni < 2; ++ni) {
            const unsigned short* p = &Bs[buf][wn * 32 + ni * 16 + l16][half * 8];
            b[ni].q[0] = *(const uint4*)p;
            b[ni].q[1] = *(const uint4*)(p + 16);
        }
#pragma unroll
        for (int mi = 0; mi < 4; ++mi)
#pragma unroll
            for (int ni = 0; ni < 2; ++ni)
                acc[mi][ni] = __builtin_amdgcn_wmma_f32_16x16x32_bf16(
                    false, a[mi].v, false, b[ni].v,
                    (short)0, acc[mi][ni], false, false);
    };

    issue_main(0);                        // prologue: tile 0 -> buf 0 (gaT/gbT -> tile 1)

    // steady state: rolled loop, 63 iterations x 2 steps; buffer parity static per step
#pragma unroll 1
    for (int it = 0; it < (KSTEPS - 2) / 2; ++it) {       // tiles 0..125
        wait_async0();                    // step 2it: tile in buf 0
        __syncthreads();
        issue_main(1);                    // tile 2it+1 -> buf 1
        compute(0);
        wait_async0();                    // step 2it+1: tile in buf 1
        __syncthreads();
        issue_main(0);                    // tile 2it+2 -> buf 0
        compute(1);
    }

    // kt = 126 (buf 0): issue tile 127 -> buf 1
    wait_async0();
    __syncthreads();
    issue_main(1);
    compute(0);

    // kt = 127 (buf 1): issue fused LoRA step (zero-padded K=32 tile) -> buf 0
    wait_async0();
    __syncthreads();
    {
        const unsigned short* ga = tL + (size_t)(mBlock + ldRow) * 32 + ldCol;
        const unsigned short* gb = WL + (size_t)(nBlock + ldRow) * 32 + ldCol;
        async_ld16(aLds,      ga);
        async_ld16(aLds + 16, ga + 8);
        async_ld16(bLds,      gb);
        async_ld16(bLds + 16, gb + 8);
    }
    compute(1);

    // kt = 128: drain — compute the LoRA step
    wait_async0();
    __syncthreads();
    compute(0);

    // epilogue: bias + store (D layout: m = v + 8*half, n = lane%16)
#pragma unroll
    for (int ni = 0; ni < 2; ++ni) {
        int n = nBlock + wn * 32 + ni * 16 + l16;
        float bv = bias[n];
#pragma unroll
        for (int mi = 0; mi < 4; ++mi) {
            int mbase = mBlock + wm * 64 + mi * 16 + half * 8;
#pragma unroll
            for (int v = 0; v < 8; ++v)
                out[(size_t)(mbase + v) * OUT_F + n] = acc[mi][ni][v] + bv;
        }
    }
}

// ---------------- launch ----------------

extern "C" void kernel_launch(void* const* d_in, const int* in_sizes, int n_in,
                              void* d_out, int out_size, void* d_ws, size_t ws_size,
                              hipStream_t stream) {
    const float* x       = (const float*)d_in[0];
    const int*   indices = (const int*)  d_in[1];
    const float* cb      = (const float*)d_in[2];
    const float* scales  = (const float*)d_in[3];
    const float* loraA   = (const float*)d_in[4];
    const float* loraB   = (const float*)d_in[5];
    const float* bias    = (const float*)d_in[6];
    float* out = (float*)d_out;

    char* ws = (char*)d_ws;
    unsigned short* W  = (unsigned short*)ws;                           // 32 MB
    unsigned short* xb = (unsigned short*)(ws + (size_t)33554432);      // 64 MB
    unsigned short* tL = (unsigned short*)(ws + (size_t)33554432 + 67108864);
    unsigned short* WL = (unsigned short*)(ws + (size_t)33554432 + 67108864 + 524288);

    dequant_kernel <<<(IN_F * OUT_F / 2) / 256, 256, 0, stream>>>(indices, cb, scales, W);
    cast_x_kernel  <<<(M_TOT * IN_F / 8) / 256, 256, 0, stream>>>(x, xb);
    lora_xA_kernel <<<(M_TOT * 16) / 256,       256, 0, stream>>>(x, loraA, tL);
    build_WL_kernel<<<(OUT_F * 32) / 256,       256, 0, stream>>>(loraB, WL);

    dim3 grid(OUT_F / BN, M_TOT / BM);   // 32 x 64 = 2048 blocks
    gemm_kernel<<<grid, 256, 0, stream>>>(xb, W, tL, WL, bias, out);
}
